// MultiHeadAttention_14302241096108
// MI455X (gfx1250) — matrix operand order repaired and checked by reading, not compiled, not run on hardware
//
#include <hip/hip_runtime.h>
#include <hip/hip_bf16.h>

// ---------------------------------------------------------------------------
// Fused MHA forward for gfx1250 (MI455X): LN -> QKV GEMM -> causal flash
// attention -> out-proj, all matmuls via v_wmma_f32_16x16x32_bf16 (wave32).
// Round 3: sched_barrier-enforced software pipelining so each K-step's loads
// issue BEFORE the previous step's WMMAs (waits become loadcnt<=N instead of
// loadcnt==0); V^T B-fragment loads hoisted above the softmax VALU block.
// ---------------------------------------------------------------------------

#define SEQ     2048
#define BATCH   2
#define HID     1024
#define NH      16
#define HD      64
#define TOK     (SEQ * BATCH)          // 4096
#define QKVN    (3 * HID)              // 3072

typedef __bf16 v16bf __attribute__((ext_vector_type(16)));
typedef float  v8f   __attribute__((ext_vector_type(8)));

struct Bytes32 { uint4 lo, hi; };      // 32B carrier for a 16-element bf16 frag

// round-to-nearest-even f32 -> bf16 (raw ushort)
__device__ __forceinline__ unsigned short f2bf(float f) {
    unsigned u = __float_as_uint(f);
    u += 0x7FFFu + ((u >> 16) & 1u);
    return (unsigned short)(u >> 16);
}

// Assemble a v16bf WMMA fragment from two 16-byte chunks (global or LDS).
__device__ __forceinline__ v16bf frag2(const void* p0, const void* p1) {
    Bytes32 t;
    t.lo = *(const uint4*)p0;
    t.hi = *(const uint4*)p1;
    return __builtin_bit_cast(v16bf, t);
}

#define WMMA_BF16(A, B, C) \
    __builtin_amdgcn_wmma_f32_16x16x32_bf16(false, (A), false, (B), (short)0, (C), false, false)

__device__ __forceinline__ void wait_lds() {
#if __has_builtin(__builtin_amdgcn_s_wait_dscnt)
    __builtin_amdgcn_s_wait_dscnt(0);
#else
    asm volatile("s_wait_dscnt 0x0" ::: "memory");
#endif
}

// Hard scheduling fence: nothing moves across (keeps prefetch loads issued
// ahead of the WMMA group so load latency overlaps matrix math).
__device__ __forceinline__ void sched_fence() {
#if __has_builtin(__builtin_amdgcn_sched_barrier)
    __builtin_amdgcn_sched_barrier(0);
#endif
}

// ---------------------------------------------------------------------------
// Kernel 1: transpose f32 [K][N] weights into bf16 [N][K] so WMMA B-operand
// per-lane reads (16 contiguous K values) are two b128 loads.
// ---------------------------------------------------------------------------
__global__ void k_transpose_bf16(const float* __restrict__ in,
                                 unsigned short* __restrict__ out,
                                 int K, int N) {
    size_t idx = (size_t)blockIdx.x * blockDim.x + threadIdx.x;
    if (idx >= (size_t)K * N) return;
    int n = (int)(idx / K);
    int k = (int)(idx % K);
    out[idx] = f2bf(in[(size_t)k * N + n]);
}

// ---------------------------------------------------------------------------
// Kernel 2: LayerNorm, one wave per token, shfl-xor reductions, bf16 out.
// inputs_q is (s,b,h) row-major == token-major with token = s*BATCH + b.
// ---------------------------------------------------------------------------
__global__ void k_layernorm(const float* __restrict__ x,
                            const float* __restrict__ gamma,
                            const float* __restrict__ beta,
                            unsigned short* __restrict__ out) {
    int token = blockIdx.x;
    int l = threadIdx.x;                       // 0..31
    const float* row = x + (size_t)token * HID;
    float s = 0.f, s2 = 0.f;
    for (int i = l; i < HID; i += 32) { float v = row[i]; s += v; s2 += v * v; }
    for (int off = 16; off > 0; off >>= 1) {
        s  += __shfl_xor(s,  off, 32);
        s2 += __shfl_xor(s2, off, 32);
    }
    float mu  = s * (1.f / HID);
    float var = s2 * (1.f / HID) - mu * mu;
    float inv = rsqrtf(var + 1e-6f);
    unsigned short* orow = out + (size_t)token * HID;
    for (int i = l; i < HID; i += 32)
        orow[i] = f2bf((row[i] - mu) * inv * gamma[i] + beta[i]);
}

// ---------------------------------------------------------------------------
// Kernel 3: QKV GEMM. A = ln_out [TOK][HID] bf16, BT = wqkvT [QKVN][HID] bf16.
// Each wave: 16 rows x 64 cols, K-loop step 32, 4 wmma per step.
// Pipeline order enforced by sched_fence: loads(k+32) || wmma(k).
// Q,K stored [b][h][s][HD]; V stored TRANSPOSED [b][h][d][s] for attention.
// ---------------------------------------------------------------------------
__global__ void k_qkv_gemm(const unsigned short* __restrict__ A,
                           const unsigned short* __restrict__ BT,
                           unsigned short* __restrict__ Qb,
                           unsigned short* __restrict__ Kb,
                           unsigned short* __restrict__ Vt) {
    const int TN = QKVN / 64;                          // 48 col-tiles
    int wid  = blockIdx.x * (blockDim.x >> 5) + (threadIdx.x >> 5);
    int lane = threadIdx.x & 31;
    int tm = wid / TN, tn = wid % TN;
    int mbase = tm * 16, nbase = tn * 64;
    int ml = lane & 15, kh = lane >> 4;

    v8f acc[4];
    for (int c = 0; c < 4; c++) { v8f z = {}; acc[c] = z; }

    const unsigned short* arow = A + (size_t)(mbase + ml) * HID;
    auto ldA = [&](int k0) {
        return frag2(arow + k0 + kh * 16, arow + k0 + kh * 16 + 8);
    };
    auto ldB = [&](int k0, int c) {
        const unsigned short* p =
            BT + (size_t)(nbase + c * 16 + ml) * HID + k0 + kh * 16;
        return frag2(p, p + 8);
    };

    // software pipeline: fragments for step k0 are loaded in step k0-32
    v16bf an = ldA(0);
    v16bf bn0 = ldB(0, 0), bn1 = ldB(0, 1), bn2 = ldB(0, 2), bn3 = ldB(0, 3);
    for (int k0 = 0; k0 < HID; k0 += 32) {
        v16bf a = an, b0 = bn0, b1 = bn1, b2 = bn2, b3 = bn3;
        int kn = k0 + 32;
        if (kn < HID) {
            an  = ldA(kn);
            bn0 = ldB(kn, 0); bn1 = ldB(kn, 1);
            bn2 = ldB(kn, 2); bn3 = ldB(kn, 3);
        }
        sched_fence();   // keep next-step loads issued before this step's wmmas
        acc[0] = WMMA_BF16(a, b0, acc[0]);
        acc[1] = WMMA_BF16(a, b1, acc[1]);
        acc[2] = WMMA_BF16(a, b2, acc[2]);
        acc[3] = WMMA_BF16(a, b3, acc[3]);
        sched_fence();
    }

    // nbase is 64-aligned -> single (t, head) per wave tile
    int t    = nbase / HID;
    int head = (nbase % HID) / HD;
    for (int c = 0; c < 4; c++)
        for (int r = 0; r < 8; r++) {
            int m  = mbase + r + 8 * kh;               // token = s*BATCH + b
            int s  = m >> 1, bt = m & 1;
            int d  = c * 16 + ml;
            unsigned short v = f2bf(acc[c][r]);
            if (t == 2) {
                // V transposed: [b][h][d][s]
                Vt[((size_t)(bt * NH + head) * HD + d) * SEQ + s] = v;
            } else {
                unsigned short* buf = (t == 0) ? Qb : Kb;
                buf[((size_t)(bt * NH + head) * SEQ + s) * HD + d] = v;
            }
        }
}

// ---------------------------------------------------------------------------
// Kernel 4: causal flash attention. One wave per (b,h,q-tile of 16 rows).
// Per 32-key step: S = Q K^T (4 wmma), online softmax, P via LDS relayout,
// O += P V (4 wmma, B-fragments contiguous from transposed V).
// V^T fragments are loaded BEFORE the softmax block so their latency
// overlaps the exp/shuffle VALU work. ctx written bf16 [token][NH*HD].
// ---------------------------------------------------------------------------
__global__ void k_attention(const unsigned short* __restrict__ Qb,
                            const unsigned short* __restrict__ Kb,
                            const unsigned short* __restrict__ Vt,
                            unsigned short* __restrict__ ctx) {
    __shared__ __align__(16) unsigned short smem[4][16 * 32];   // P tile / wave

    int wslot = threadIdx.x >> 5;
    int wid   = blockIdx.x * 4 + wslot;
    int lane  = threadIdx.x & 31;
    int bh = wid >> 7;                  // 128 q-tiles per (b,h)
    int qt = wid & 127;
    int b  = bh >> 4, h = bh & 15;
    int ml = lane & 15, kh = lane >> 4;
    int qbase = qt * 16;

    const unsigned short* Qp  = Qb + (size_t)bh * SEQ * HD;
    const unsigned short* Kp  = Kb + (size_t)bh * SEQ * HD;
    const unsigned short* Vtp = Vt + (size_t)bh * HD * SEQ;    // [d][s]

    // Q A-fragments for K-dim 0..31 and 32..63 (held for whole key loop)
    const unsigned short* qrow = Qp + (size_t)(qbase + ml) * HD;
    v16bf qa0 = frag2(qrow + kh * 16,      qrow + kh * 16 + 8);
    v16bf qa1 = frag2(qrow + 32 + kh * 16, qrow + 32 + kh * 16 + 8);

    v8f o[4];
    for (int c = 0; c < 4; c++) { v8f z = {}; o[c] = z; }
    float mst[8], lst[8];
    for (int r = 0; r < 8; r++) { mst[r] = -INFINITY; lst[r] = 0.f; }

    const float sc = 0.125f;                    // 1/sqrt(HD)
    unsigned short* pl = smem[wslot];
    int nsteps = (qbase >> 5) + 1;              // kb <= qbase (causal)

    for (int st = 0; st < nsteps; st++) {
        int kb = st * 32;

        // prefetch next key-step's K rows and V^T spans into cache
        if (st + 1 < nsteps) {
            __builtin_prefetch(Kp + (size_t)(kb + 32 + lane) * HD, 0, 3);
            __builtin_prefetch(Vtp + (size_t)lane * SEQ + kb + 32, 0, 3);
            __builtin_prefetch(Vtp + (size_t)(32 + lane) * SEQ + kb + 32, 0, 3);
        }

        v8f s0 = {}, s1 = {};
        // S tile 0: keys kb..kb+15 ; B[k][n] = K[kb+n][k] (contig K rows)
        {
            const unsigned short* kr = Kp + (size_t)(kb + ml) * HD;
            v16bf b0 = frag2(kr + kh * 16,      kr + kh * 16 + 8);
            v16bf b1 = frag2(kr + 32 + kh * 16, kr + 32 + kh * 16 + 8);
            s0 = WMMA_BF16(qa0, b0, s0);
            s0 = WMMA_BF16(qa1, b1, s0);
        }
        // S tile 1: keys kb+16..kb+31
        {
            const unsigned short* kr = Kp + (size_t)(kb + 16 + ml) * HD;
            v16bf b0 = frag2(kr + kh * 16,      kr + kh * 16 + 8);
            v16bf b1 = frag2(kr + 32 + kh * 16, kr + 32 + kh * 16 + 8);
            s1 = WMMA_BF16(qa0, b0, s1);
            s1 = WMMA_BF16(qa1, b1, s1);
        }

        // V^T B-fragments issued now; latency overlaps softmax VALU below.
        v16bf vb0, vb1, vb2, vb3;
        {
            const unsigned short* vr0 = Vtp + (size_t)(ml)      * SEQ + kb + kh * 16;
            const unsigned short* vr1 = Vtp + (size_t)(16 + ml) * SEQ + kb + kh * 16;
            const unsigned short* vr2 = Vtp + (size_t)(32 + ml) * SEQ + kb + kh * 16;
            const unsigned short* vr3 = Vtp + (size_t)(48 + ml) * SEQ + kb + kh * 16;
            vb0 = frag2(vr0, vr0 + 8);
            vb1 = frag2(vr1, vr1 + 8);
            vb2 = frag2(vr2, vr2 + 8);
            vb3 = frag2(vr3, vr3 + 8);
        }
        sched_fence();   // don't sink the V^T loads into the softmax block

        // scale + causal mask + online softmax (row = r + 8*kh, col = ml)
        for (int r = 0; r < 8; r++) {
            int qrowi = qbase + r + 8 * kh;
            float v0 = s0[r] * sc; if (kb + ml      > qrowi) v0 = -1e30f;
            float v1 = s1[r] * sc; if (kb + 16 + ml > qrowi) v1 = -1e30f;
            float rmax = fmaxf(v0, v1);
            for (int off = 1; off < 16; off <<= 1)
                rmax = fmaxf(rmax, __shfl_xor(rmax, off, 32));
            float mnew = fmaxf(mst[r], rmax);
            float corr = __expf(mst[r] - mnew);
            float p0 = __expf(v0 - mnew);
            float p1 = __expf(v1 - mnew);
            float rs = p0 + p1;
            for (int off = 1; off < 16; off <<= 1)
                rs += __shfl_xor(rs, off, 32);
            lst[r] = lst[r] * corr + rs;
            mst[r] = mnew;
            for (int c = 0; c < 4; c++) o[c][r] *= corr;
            int row = r + 8 * kh;
            pl[row * 32 + ml]      = f2bf(p0);
            pl[row * 32 + 16 + ml] = f2bf(p1);
        }
        wait_lds();

        // reload P (16x32) from LDS in A-fragment layout
        const unsigned short* prow = pl + ml * 32;
        v16bf pa = frag2(prow + kh * 16, prow + kh * 16 + 8);

        // O += P @ V ; B[k][n] = V[kb+k][c*16+n] = Vt[c*16+n][kb+k]
        o[0] = WMMA_BF16(pa, vb0, o[0]);
        o[1] = WMMA_BF16(pa, vb1, o[1]);
        o[2] = WMMA_BF16(pa, vb2, o[2]);
        o[3] = WMMA_BF16(pa, vb3, o[3]);
    }

    // epilogue: normalize, write ctx[token][h*HD + d] as bf16
    for (int r = 0; r < 8; r++) {
        int s = qbase + r + 8 * kh;
        size_t token = (size_t)s * BATCH + b;
        float invl = 1.f / lst[r];
        for (int c = 0; c < 4; c++)
            ctx[token * (NH * HD) + h * HD + c * 16 + ml] = f2bf(o[c][r] * invl);
    }
}

// ---------------------------------------------------------------------------
// Kernel 5: output projection. ctx [TOK][HID] bf16 @ woutT -> out f32.
// Same sched_fence pipelining as k_qkv_gemm.
// ---------------------------------------------------------------------------
__global__ void k_out_gemm(const unsigned short* __restrict__ A,
                           const unsigned short* __restrict__ BT,
                           float* __restrict__ out) {
    const int TN = HID / 64;                           // 16 col-tiles
    int wid  = blockIdx.x * (blockDim.x >> 5) + (threadIdx.x >> 5);
    int lane = threadIdx.x & 31;
    int tm = wid / TN, tn = wid % TN;
    int mbase = tm * 16, nbase = tn * 64;
    int ml = lane & 15, kh = lane >> 4;

    v8f acc[4];
    for (int c = 0; c < 4; c++) { v8f z = {}; acc[c] = z; }

    const unsigned short* arow = A + (size_t)(mbase + ml) * HID;
    auto ldA = [&](int k0) {
        return frag2(arow + k0 + kh * 16, arow + k0 + kh * 16 + 8);
    };
    auto ldB = [&](int k0, int c) {
        const unsigned short* p =
            BT + (size_t)(nbase + c * 16 + ml) * HID + k0 + kh * 16;
        return frag2(p, p + 8);
    };

    v16bf an = ldA(0);
    v16bf bn0 = ldB(0, 0), bn1 = ldB(0, 1), bn2 = ldB(0, 2), bn3 = ldB(0, 3);
    for (int k0 = 0; k0 < HID; k0 += 32) {
        v16bf a = an, b0 = bn0, b1 = bn1, b2 = bn2, b3 = bn3;
        int kn = k0 + 32;
        if (kn < HID) {
            an  = ldA(kn);
            bn0 = ldB(kn, 0); bn1 = ldB(kn, 1);
            bn2 = ldB(kn, 2); bn3 = ldB(kn, 3);
        }
        sched_fence();
        acc[0] = WMMA_BF16(a, b0, acc[0]);
        acc[1] = WMMA_BF16(a, b1, acc[1]);
        acc[2] = WMMA_BF16(a, b2, acc[2]);
        acc[3] = WMMA_BF16(a, b3, acc[3]);
        sched_fence();
    }
    for (int c = 0; c < 4; c++)
        for (int r = 0; r < 8; r++) {
            int m = mbase + r + 8 * kh;
            out[(size_t)m * HID + nbase + c * 16 + ml] = acc[c][r];
        }
}

// ---------------------------------------------------------------------------
extern "C" void kernel_launch(void* const* d_in, const int* in_sizes, int n_in,
                              void* d_out, int out_size, void* d_ws, size_t ws_size,
                              hipStream_t stream) {
    const float* x     = (const float*)d_in[0];   // (SEQ,BATCH,HID)
    const float* gamma = (const float*)d_in[1];   // (HID)
    const float* beta  = (const float*)d_in[2];   // (HID)
    const float* wqkv  = (const float*)d_in[3];   // (HID,3,NH*HD)
    const float* wout  = (const float*)d_in[4];   // (NH*HD,HID)
    float* out = (float*)d_out;

    char* ws = (char*)d_ws;
    size_t off = 0;
    auto alloc = [&](size_t bytes) -> void* {
        void* p = ws + off;
        off += (bytes + 255) & ~(size_t)255;
        return p;
    };
    unsigned short* lnq   = (unsigned short*)alloc((size_t)TOK  * HID * 2);  // 8 MB
    unsigned short* wqkvT = (unsigned short*)alloc((size_t)QKVN * HID * 2);  // 6 MB
    unsigned short* woutT = (unsigned short*)alloc((size_t)HID  * HID * 2);  // 2 MB
    unsigned short* Qb    = (unsigned short*)alloc((size_t)TOK  * HID * 2);  // 8 MB
    unsigned short* Kb    = (unsigned short*)alloc((size_t)TOK  * HID * 2);  // 8 MB
    unsigned short* Vt    = (unsigned short*)alloc((size_t)TOK  * HID * 2);  // 8 MB (transposed)
    unsigned short* ctx   = (unsigned short*)alloc((size_t)TOK  * HID * 2);  // 8 MB

    // 1) weight transposes (independent)
    {
        size_t n1 = (size_t)HID * QKVN;
        k_transpose_bf16<<<(unsigned)((n1 + 255) / 256), 256, 0, stream>>>(wqkv, wqkvT, HID, QKVN);
        size_t n2 = (size_t)HID * HID;
        k_transpose_bf16<<<(unsigned)((n2 + 255) / 256), 256, 0, stream>>>(wout, woutT, HID, HID);
    }
    // 2) layernorm
    k_layernorm<<<TOK, 32, 0, stream>>>(x, gamma, beta, lnq);
    // 3) QKV GEMM: (TOK/16)*(QKVN/64) = 256*48 = 12288 waves, 4 waves/block
    k_qkv_gemm<<<12288 / 4, 128, 0, stream>>>(lnq, wqkvT, Qb, Kb, Vt);
    // 4) attention: 32 (b,h) * 128 q-tiles = 4096 waves, 4 waves/block
    k_attention<<<4096 / 4, 128, 0, stream>>>(Qb, Kb, Vt, ctx);
    // 5) out projection: 256*16 = 4096 waves
    k_out_gemm<<<4096 / 4, 128, 0, stream>>>(ctx, woutT, out);
}
